// ChannelCrissCrossAttention_71073118814791
// MI455X (gfx1250) — matrix operand order repaired and checked by reading, not compiled, not run on hardware
//
#include <hip/hip_runtime.h>
#include <stdint.h>

// ---------------- types for WMMA operands ----------------
typedef __bf16 bf16x16 __attribute__((ext_vector_type(16)));
typedef float  f32x8   __attribute__((ext_vector_type(8)));
typedef unsigned int u32x4 __attribute__((ext_vector_type(4)));

union BF16V {
    bf16x16 v;
    u32x4   q[2];
};

// ---------------- problem constants ----------------
#define Bn   8
#define INC  128
#define CC   16
#define Vd   64
#define Td   64
#define Sd   (Vd*Td)            // 4096
#define KQ   (9*INC)            // 1152  (K for qkv convs)
#define KR   160                // 144 padded to 160 (K for reverse convs)
#define QSZ  ((size_t)Bn*CC*Sd)         // one q/k/v tensor, f32 elements

// padded (halo) geometry
#define XP_W   66                        // Td+2 cols
#define XP_H   66                        // Vd+2 rows
#define XPLANE ((size_t)XP_H*XP_W*INC)   // 557,568 bf16 elems per image
#define AP_W   66                        // Td+2 cols
#define AP_H   68                        // Vd+4 rows (dv=+2 from padded-K chunk stays legal)
#define APLANE ((size_t)AP_H*AP_W*CC)    // 71,808 bf16 elems per image
#define APSZ   ((size_t)Bn*APLANE)       // one attention-out tensor (padded)

__device__ __forceinline__ unsigned short f2bf(float f) {
    unsigned int u = __float_as_uint(f);
    u += 0x7fffu + ((u >> 16) & 1u);          // round-to-nearest-even
    return (unsigned short)(u >> 16);
}

// ---------------- prep: x f32 NCHW -> bf16 NHWC with zero halo ----------------
__global__ __launch_bounds__(256) void cast_x_kernel(const float* __restrict__ x,
                                                     unsigned short* __restrict__ xb) {
    int i = blockIdx.x * 256 + threadIdx.x;           // over B*INC*V*T
    if (i >= Bn * INC * Sd) return;
    int t  = i & 63;
    int r  = i >> 6;
    int vv = r & 63;  r >>= 6;
    int ci = r & 127; int b = r >> 7;
    size_t o = ((size_t)b * (XP_H * XP_W) + (size_t)(vv + 1) * XP_W + (t + 1)) * INC + ci;
    xb[o] = f2bf(x[i]);
}

// ---------------- prep: qkv weights -> bf16 [z][m][kpos*128+ci] ----------------
__global__ __launch_bounds__(256) void prep_wqkv_kernel(const float* __restrict__ wq,
                                                        const float* __restrict__ wk,
                                                        const float* __restrict__ wv,
                                                        unsigned short* __restrict__ Wt) {
    int i = blockIdx.x * 256 + threadIdx.x;           // over 3*16*1152
    if (i >= 3 * CC * KQ) return;
    int K = i % KQ;
    int r = i / KQ;
    int m = r % CC;
    int z = r / CC;
    int kpos = K >> 7;
    int ci   = K & 127;
    const float* w = (z == 0) ? wq : (z == 1) ? wk : wv;
    Wt[i] = f2bf(w[(m * INC + ci) * 9 + kpos]);
}

// ---------------- prep: reverse weights -> bf16 [z][m][K], K padded to 160 ----------------
__global__ __launch_bounds__(256) void prep_wrev_kernel(const float* __restrict__ wcr,
                                                        const float* __restrict__ wvr,
                                                        const float* __restrict__ wtr,
                                                        unsigned short* __restrict__ Wr) {
    int i = blockIdx.x * 256 + threadIdx.x;           // over 3*128*160
    if (i >= 3 * INC * KR) return;
    int K = i % KR;
    int r = i / KR;
    int m = r % INC;
    int z = r / INC;
    int kpos = K >> 4;
    int ci   = K & 15;
    const float* w = (z == 0) ? wcr : (z == 1) ? wvr : wtr;
    Wr[i] = (kpos < 9) ? f2bf(w[(m * CC + ci) * 9 + kpos]) : (unsigned short)0;
}

// ---------------- conv q/k/v : implicit GEMM, M=16(C), K=1152, N=16 per wave ----------------
// branchless: xb has a zero halo, every address is legal
__global__ __launch_bounds__(256) void conv_qkv_kernel(const unsigned short* __restrict__ xb,
                                                       const unsigned short* __restrict__ Wt,
                                                       const float* __restrict__ bq,
                                                       const float* __restrict__ bk,
                                                       const float* __restrict__ bv,
                                                       float* __restrict__ qkv) {
    const int z    = blockIdx.z;                      // 0=q 1=k 2=v
    const int b    = blockIdx.y;
    const int w    = threadIdx.x >> 5;
    const int lane = threadIdx.x & 31;
    const int g    = lane >> 4;                       // lane group (K half)
    const int Ml   = lane & 15;                       // A-row (M) and B-col (N)
    const int s0   = blockIdx.x * 128 + w * 16;       // 16 spatial positions, one v row
    const int v0   = s0 >> 6;
    const int t0   = s0 & 63;

    const unsigned short* Wz  = Wt + (size_t)z * CC * KQ;
    const unsigned short* xbb = xb + (size_t)b * XPLANE;

    f32x8 acc = {0.f,0.f,0.f,0.f,0.f,0.f,0.f,0.f};

    for (int kpos = 0; kpos < 9; ++kpos) {
        const int dv = kpos / 3 - 1;
        const int dt = kpos % 3 - 1;
        // halo-shifted source pixel, always in-bounds
        const unsigned short* xrow =
            xbb + ((size_t)(v0 + dv + 1) * XP_W + (t0 + Ml + dt + 1)) * INC;
        __builtin_prefetch((const void*)xrow, 0, 0);
        #pragma unroll
        for (int cb = 0; cb < 4; ++cb) {
            const int K0 = kpos * 128 + cb * 32;
            BF16V A, Bm;
            const unsigned short* arow = Wz + (size_t)Ml * KQ + K0 + g * 16;
            A.q[0] = *(const u32x4*)(arow);
            A.q[1] = *(const u32x4*)(arow + 8);
            const unsigned short* bp = xrow + cb * 32 + g * 16;
            Bm.q[0] = *(const u32x4*)(bp);
            Bm.q[1] = *(const u32x4*)(bp + 8);
            acc = __builtin_amdgcn_wmma_f32_16x16x32_bf16(
                      false, A.v, false, Bm.v, (short)0, acc, false, false);
        }
    }

    const float* bias = (z == 0) ? bq : (z == 1) ? bk : bv;
    float* outz = qkv + (size_t)z * QSZ + (size_t)b * CC * Sd;
    #pragma unroll
    for (int r = 0; r < 8; ++r) {
        const int Mp = r + 8 * g;                     // C/D layout: vgpr r, lane-half g
        outz[(size_t)Mp * Sd + s0 + Ml] = acc[r] + bias[Mp];
    }
}

// ---------------- criss-cross attention core (softmax over C+V+T) ----------------
__global__ __launch_bounds__(256) void attn_kernel(const float* __restrict__ qkv,
                                                   unsigned short* __restrict__ attb) {
    const float* qb = qkv;
    const float* kb = qkv + QSZ;
    const float* vb = qkv + 2 * QSZ;
    const int b = blockIdx.x >> 4;
    const int c = blockIdx.x & 15;

    __shared__ float ks[Sd];                          // k[b][c][:,:]
    __shared__ float vs[Sd];                          // v[b][c][:,:]
    const size_t base = ((size_t)(b * CC + c)) * Sd;
    for (int i = threadIdx.x; i < Sd; i += 256) {
        ks[i] = kb[base + i];
        vs[i] = vb[base + i];
    }
    __syncthreads();

    const int t     = threadIdx.x & 63;
    const int vrow0 = blockIdx.y * 16 + (threadIdx.x >> 6) * 4;

    for (int vi = 0; vi < 4; ++vi) {
        const int vr = vrow0 + vi;
        const int s  = vr * 64 + t;
        const float qs = qb[base + s];

        float kc[16], vc[16];
        #pragma unroll
        for (int j = 0; j < 16; ++j) {
            kc[j] = kb[((size_t)(b * CC + j)) * Sd + s];
            vc[j] = vb[((size_t)(b * CC + j)) * Sd + s];
        }
        // all energies are qs * k  ->  max(e) from min/max of k
        float kmax = -3.402823466e38f, kmin = 3.402823466e38f;
        #pragma unroll
        for (int j = 0; j < 16; ++j) { kmax = fmaxf(kmax, kc[j]); kmin = fminf(kmin, kc[j]); }
        for (int v2 = 0; v2 < 64; ++v2) { float x = ks[v2 * 64 + t]; kmax = fmaxf(kmax, x); kmin = fminf(kmin, x); }
        for (int t2 = 0; t2 < 64; ++t2) { float x = ks[vr * 64 + t2]; kmax = fmaxf(kmax, x); kmin = fminf(kmin, x); }
        const float m = fmaxf(qs * kmax, qs * kmin);

        float Z = 0.f, oc = 0.f, ov = 0.f, ot = 0.f;
        #pragma unroll
        for (int j = 0; j < 16; ++j) { float w = __expf(qs * kc[j] - m); Z += w; oc += w * vc[j]; }
        for (int v2 = 0; v2 < 64; ++v2) { float w = __expf(qs * ks[v2 * 64 + t] - m); Z += w; ov += w * vs[v2 * 64 + t]; }
        for (int t2 = 0; t2 < 64; ++t2) { float w = __expf(qs * ks[vr * 64 + t2] - m); Z += w; ot += w * vs[vr * 64 + t2]; }
        const float rz = 1.0f / Z;

        // halo-padded NHWC bf16 outputs
        const size_t o = ((size_t)b * (AP_H * AP_W) + (size_t)(vr + 1) * AP_W + (t + 1)) * CC + c;
        attb[o]            = f2bf(oc * rz);
        attb[APSZ + o]     = f2bf(ov * rz);
        attb[2 * APSZ + o] = f2bf(ot * rz);
    }
}

// ---------------- reverse convs + residual: M=128, K=3x160, N=16 per block ----------------
// branchless: att buffers have zero halos; padded-K rows have zero A weights
__global__ __launch_bounds__(256) void conv_rev_kernel(const unsigned short* __restrict__ attb,
                                                       const unsigned short* __restrict__ Wr,
                                                       const float* __restrict__ bcr,
                                                       const float* __restrict__ bvr,
                                                       const float* __restrict__ btr,
                                                       const float* __restrict__ gamma,
                                                       const float* __restrict__ x,
                                                       float* __restrict__ out) {
    const int b    = blockIdx.y;
    const int w    = threadIdx.x >> 5;
    const int lane = threadIdx.x & 31;
    const int g    = lane >> 4;
    const int Ml   = lane & 15;
    const int m0   = w * 16;                          // wave's M tile
    const int s0   = blockIdx.x * 16;                 // block's N tile (16 spatial)
    const int v0   = s0 >> 6;
    const int t0   = s0 & 63;

    f32x8 acc = {0.f,0.f,0.f,0.f,0.f,0.f,0.f,0.f};

    for (int ten = 0; ten < 3; ++ten) {
        const unsigned short* src = attb + (size_t)ten * APSZ + (size_t)b * APLANE;
        const unsigned short* Wz  = Wr + (size_t)ten * INC * KR;
        #pragma unroll
        for (int ch = 0; ch < 5; ++ch) {
            BF16V A, Bm;
            const unsigned short* arow = Wz + (size_t)(m0 + Ml) * KR + ch * 32 + g * 16;
            A.q[0] = *(const u32x4*)(arow);
            A.q[1] = *(const u32x4*)(arow + 8);

            const int kpos = 2 * ch + g;              // lane half g owns one kpos block
            const int dv = kpos / 3 - 1;              // kpos==9 -> dv=+2: legal (AP_H=68), A rows are zero
            const int dt = kpos % 3 - 1;
            const unsigned short* bp =
                src + ((size_t)(v0 + dv + 1) * AP_W + (t0 + Ml + dt + 1)) * CC;
            Bm.q[0] = *(const u32x4*)(bp);
            Bm.q[1] = *(const u32x4*)(bp + 8);

            acc = __builtin_amdgcn_wmma_f32_16x16x32_bf16(
                      false, A.v, false, Bm.v, (short)0, acc, false, false);
        }
    }

    const float gm = gamma[0];
    #pragma unroll
    for (int r = 0; r < 8; ++r) {
        const int Mp = m0 + r + 8 * g;
        const size_t o = ((size_t)(b * INC + Mp)) * Sd + s0 + Ml;
        out[o] = gm * (acc[r] + bcr[Mp] + bvr[Mp] + btr[Mp]) + x[o];
    }
}

// ---------------- host launcher ----------------
extern "C" void kernel_launch(void* const* d_in, const int* in_sizes, int n_in,
                              void* d_out, int out_size, void* d_ws, size_t ws_size,
                              hipStream_t stream) {
    const float* x     = (const float*)d_in[0];
    const float* wq    = (const float*)d_in[1];
    const float* bq    = (const float*)d_in[2];
    const float* wk    = (const float*)d_in[3];
    const float* bk    = (const float*)d_in[4];
    const float* wv    = (const float*)d_in[5];
    const float* bv    = (const float*)d_in[6];
    const float* wcr   = (const float*)d_in[7];
    const float* bcr   = (const float*)d_in[8];
    const float* wvr   = (const float*)d_in[9];
    const float* bvr   = (const float*)d_in[10];
    const float* wtr   = (const float*)d_in[11];
    const float* btr   = (const float*)d_in[12];
    const float* gamma = (const float*)d_in[13];
    float* out = (float*)d_out;

    // workspace carve-up
    const size_t xb_bytes  = (size_t)Bn * XPLANE * 2;        // 8,921,088
    const size_t wt_bytes  = (size_t)3 * CC * KQ * 2;        // 110,592
    const size_t wr_bytes  = (size_t)3 * INC * KR * 2;       // 122,880
    const size_t qkv_bytes = (size_t)3 * QSZ * sizeof(float);// 6,291,456
    const size_t att_bytes = (size_t)3 * APSZ * 2;           // 3,446,784

    char* p = (char*)d_ws;
    unsigned short* xb  = (unsigned short*)p; p += xb_bytes;
    unsigned short* Wt  = (unsigned short*)p; p += wt_bytes;
    unsigned short* Wr  = (unsigned short*)p; p += wr_bytes;
    float*          qkv = (float*)p;          p += qkv_bytes;
    unsigned short* att = (unsigned short*)p; p += att_bytes;

    // zero-fill the halo'd buffers (capture-safe async memsets)
    hipMemsetAsync(xb, 0, xb_bytes, stream);
    hipMemsetAsync(att, 0, att_bytes, stream);

    const int nx = Bn * INC * Sd;                           // 4,194,304
    cast_x_kernel<<<(nx + 255) / 256, 256, 0, stream>>>(x, xb);

    const int nwq = 3 * CC * KQ;                            // 55,296
    prep_wqkv_kernel<<<(nwq + 255) / 256, 256, 0, stream>>>(wq, wk, wv, Wt);

    const int nwr = 3 * INC * KR;                           // 61,440
    prep_wrev_kernel<<<(nwr + 255) / 256, 256, 0, stream>>>(wcr, wvr, wtr, Wr);

    conv_qkv_kernel<<<dim3(Sd / 128, Bn, 3), 256, 0, stream>>>(xb, Wt, bq, bk, bv, qkv);

    attn_kernel<<<dim3(Bn * CC, 4), 256, 0, stream>>>(qkv, att);

    conv_rev_kernel<<<dim3(Sd / 16, Bn), 256, 0, stream>>>(att, Wr, bcr, bvr, btr, gamma, x, out);
}
